// GraphNetworkBlock_50044958933334
// MI455X (gfx1250) — matrix operand order repaired and checked by reading, not compiled, not run on hardware
//
#include <hip/hip_runtime.h>

#define N_NODES 50000
#define N_EDGES 800000
#define XS 200   // LDS row stride (halfwords) for K=192 tiles (padded, bank-conflict free)
#define HS 136   // LDS row stride (halfwords) for K=128 tiles

typedef __attribute__((ext_vector_type(16))) __bf16 bf16x16;
typedef __attribute__((ext_vector_type(8)))  float  floatx8;

__device__ __forceinline__ unsigned short f32_bf16(float f) {
  unsigned u = __float_as_uint(f);
  u += 0x7FFFu + ((u >> 16) & 1u);   // round-to-nearest-even
  return (unsigned short)(u >> 16);
}

__device__ __forceinline__ unsigned pack_bf16x2(float a, float b) {
  return (unsigned)f32_bf16(a) | ((unsigned)f32_bf16(b) << 16);
}

union BfFrag { bf16x16 v; uint4 q[2]; unsigned u[8]; };

// A fragment: 16x32 bf16 (ISA 7.12.2 "16-bit A-Matrix 16x32").
// lane L: M = L&15 ; lanes 16-31 use K base +8 ; u[v<4]=K(2v,2v+1), u[v>=4]=K(16+2(v-4),..)
// Both halves are contiguous 16-byte runs -> two ds_load_b128.
__device__ __forceinline__ bf16x16 load_a_frag(const unsigned short* X, int stride,
                                               int row0, int k0) {
  int lane = threadIdx.x & 31;
  const unsigned short* base = X + (row0 + (lane & 15)) * stride + k0 + ((lane >> 4) << 3);
  BfFrag f;
  f.q[0] = *(const uint4*)(base);        // halfwords +0..7
  f.q[1] = *(const uint4*)(base + 16);   // halfwords +16..23
  return f.v;
}

// B fragment: 32x16 bf16 from transposed (N-major) weight tile in LDS.
// lane L: N = L&15 ; lanes 16-31 use K base +8 ; u[v<4]=K(2v,2v+1), u[v>=4]=K(16+2(v-4),..)
// Both halves are contiguous 16-byte runs -> two ds_load_b128.
__device__ __forceinline__ bf16x16 load_b_frag(const unsigned short* Wt, int stride,
                                               int n0, int k0) {
  int lane = threadIdx.x & 31;
  const unsigned short* base = Wt + (n0 + (lane & 15)) * stride + k0 + ((lane >> 4) << 3);
  BfFrag f;
  f.q[0] = *(const uint4*)(base);        // halfwords +0..7
  f.q[1] = *(const uint4*)(base + 16);   // halfwords +16..23
  return f.v;
}

__device__ __forceinline__ floatx8 wmma_bf16(bf16x16 a, bf16x16 b, floatx8 c) {
  return __builtin_amdgcn_wmma_f32_16x16x32_bf16(false, a, false, b, (short)0, c,
                                                 false, false);
}

// ---------------------------------------------------------------------------
// Prep: zero the receiver-aggregation buffer (aliases node-output region of
// d_out) and convert all 4 weight matrices to bf16, transposed to N-major.
// ---------------------------------------------------------------------------
__global__ __launch_bounds__(256) void gn_prep_kernel(
    const float* __restrict__ eW1, const float* __restrict__ eW2,
    const float* __restrict__ nW1, const float* __restrict__ nW2,
    unsigned short* __restrict__ eW1t, unsigned short* __restrict__ eW2t,
    unsigned short* __restrict__ nW1t, unsigned short* __restrict__ nW2t,
    float* __restrict__ agg) {
  long t = (long)blockIdx.x * blockDim.x + threadIdx.x;
  long stride = (long)gridDim.x * blockDim.x;
  for (long i = t; i < (long)N_NODES * 16; i += stride)   // float4-wide zero
    ((float4*)agg)[i] = float4{0.f, 0.f, 0.f, 0.f};
  for (long i = t; i < 128L * 192; i += stride) {
    long n = i / 192, k = i - n * 192;
    eW1t[i] = f32_bf16(eW1[k * 128 + n]);     // eW1: [192,128]
  }
  for (long i = t; i < 64L * 128; i += stride) {
    long n = i >> 7, k = i & 127;
    eW2t[i] = f32_bf16(eW2[k * 64 + n]);      // eW2: [128,64]
  }
  for (long i = t; i < 128L * 128; i += stride) {
    long n = i >> 7, k = i & 127;
    nW1t[i] = f32_bf16(nW1[k * 128 + n]);     // nW1: [128,128]
  }
  for (long i = t; i < 64L * 128; i += stride) {
    long n = i >> 7, k = i & 127;
    nW2t[i] = f32_bf16(nW2[k * 64 + n]);      // nW2: [128,64]
  }
}

// ---------------------------------------------------------------------------
// Edge kernel: 128 edges/block, 8 waves. Gather [recv|send|edge_attr] -> LDS
// bf16, MLP 192->128 (ReLU) ->64 (ReLU) -> LayerNorm, write edge_new and
// atomically scatter into receiver aggregation buffer.
// ---------------------------------------------------------------------------
__global__ __launch_bounds__(256) void gn_edge_kernel(
    const float* __restrict__ edge_attr, const float* __restrict__ node_attr,
    const int* __restrict__ edge_index,
    const unsigned short* __restrict__ W1t, const unsigned short* __restrict__ W2t,
    const float* __restrict__ b1, const float* __restrict__ b2,
    const float* __restrict__ gamma, const float* __restrict__ beta,
    float* __restrict__ edge_out, float* agg) {
  extern __shared__ char smem[];
  unsigned short* Xs  = (unsigned short*)smem;      // 128 x XS
  unsigned short* W1s = Xs + 128 * XS;              // 128(N) x XS(K=192)
  unsigned short* Hs  = W1s + 128 * XS;             // 128 x HS
  unsigned short* W2s = Hs + 128 * HS;              // 64(N) x HS(K=128)
  int* rcv = (int*)(W2s + 64 * HS);                 // 128
  int* snd = rcv + 128;                             // 128

  const int tid = threadIdx.x;
  const long e0 = (long)blockIdx.x * 128;

  for (int r = tid; r < 128; r += 256) {
    long e = e0 + r;
    snd[r] = edge_index[e];                // row = senders
    rcv[r] = edge_index[N_EDGES + e];      // col = receivers
  }
  // weights -> LDS, 16 bytes at a time (both sides 16B-aligned)
  for (int i = tid; i < 128 * 24; i += 256) {       // 192/8 = 24 octets/row
    int n = i / 24, k8 = (i - n * 24) * 8;
    *(uint4*)(W1s + n * XS + k8) = *(const uint4*)(W1t + n * 192 + k8);
  }
  for (int i = tid; i < 64 * 16; i += 256) {        // 128/8 = 16 octets/row
    int n = i >> 4, k8 = (i & 15) * 8;
    *(uint4*)(W2s + n * HS + k8) = *(const uint4*)(W2t + n * 128 + k8);
  }
  __syncthreads();

  {  // gather: 2 threads per row, 96 cols each, float4 loads + packed stores
    int r = tid >> 1, h = tid & 1;
    const float* rv = node_attr + (long)rcv[r] * 64;
    const float* sv = node_attr + (long)snd[r] * 64;
    const float* ea = edge_attr + (e0 + r) * 64;
    unsigned short* xr = Xs + r * XS;
#pragma unroll 1
    for (int c = h * 96; c < h * 96 + 96; c += 4) {
      const float* src = (c < 64) ? (rv + c) : (c < 128 ? (sv + c - 64) : (ea + c - 128));
      float4 f = *(const float4*)src;
      uint2 p;
      p.x = pack_bf16x2(f.x, f.y);
      p.y = pack_bf16x2(f.z, f.w);
      *(uint2*)(xr + c) = p;
    }
  }
  __syncthreads();

  const int wave = tid >> 5;
  const int lane = tid & 31;
  const int row0 = wave << 4;
  const int nl = lane & 15;
  const int mo = (lane >> 4) << 3;

  // Layer 1: [16,192] @ [192,128]
#pragma unroll 1
  for (int nt = 0; nt < 8; ++nt) {
    floatx8 acc = {0.f, 0.f, 0.f, 0.f, 0.f, 0.f, 0.f, 0.f};
#pragma unroll
    for (int kc = 0; kc < 6; ++kc) {
      bf16x16 a = load_a_frag(Xs, XS, row0, kc * 32);
      bf16x16 b = load_b_frag(W1s, XS, nt * 16, kc * 32);
      acc = wmma_bf16(a, b, acc);
    }
    int col = nt * 16 + nl;
    float bias = b1[col];
#pragma unroll
    for (int r = 0; r < 8; ++r)
      Hs[(row0 + r + mo) * HS + col] = f32_bf16(fmaxf(acc[r] + bias, 0.f));
  }
  __syncthreads();

  // Layer 2: [16,128] @ [128,64]
  floatx8 cacc[4];
#pragma unroll
  for (int nt = 0; nt < 4; ++nt)
    cacc[nt] = floatx8{0.f, 0.f, 0.f, 0.f, 0.f, 0.f, 0.f, 0.f};
#pragma unroll
  for (int kc = 0; kc < 4; ++kc) {
    bf16x16 a = load_a_frag(Hs, HS, row0, kc * 32);
#pragma unroll
    for (int nt = 0; nt < 4; ++nt) {
      bf16x16 b = load_b_frag(W2s, HS, nt * 16, kc * 32);
      cacc[nt] = wmma_bf16(a, b, cacc[nt]);
    }
  }

  float x[4][8], g[4], bt[4];
#pragma unroll
  for (int nt = 0; nt < 4; ++nt) {
    int col = nt * 16 + nl;
    float bias = b2[col];
    g[nt] = gamma[col];
    bt[nt] = beta[col];
#pragma unroll
    for (int r = 0; r < 8; ++r) x[nt][r] = fmaxf(cacc[nt][r] + bias, 0.f);
  }

  // LayerNorm: row M = row0 + r + mo lives in one 16-lane half across 4 tiles
#pragma unroll 1
  for (int r = 0; r < 8; ++r) {
    float s = x[0][r] + x[1][r] + x[2][r] + x[3][r];
    float q = x[0][r] * x[0][r] + x[1][r] * x[1][r] +
              x[2][r] * x[2][r] + x[3][r] * x[3][r];
#pragma unroll
    for (int m = 1; m < 16; m <<= 1) {
      s += __shfl_xor(s, m, 16);
      q += __shfl_xor(q, m, 16);
    }
    float mean = s * (1.f / 64.f);
    float var  = q * (1.f / 64.f) - mean * mean;
    float inv  = rsqrtf(var + 1e-5f);
    int rl = row0 + r + mo;
    float* orow = edge_out + (e0 + rl) * 64;
    float* arow = agg + (long)rcv[rl] * 64;
#pragma unroll
    for (int nt = 0; nt < 4; ++nt) {
      float y = (x[nt][r] - mean) * inv * g[nt] + bt[nt];
      orow[nt * 16 + nl] = y;
      atomicAdd(arow + nt * 16 + nl, y);   // segment_sum over receivers
    }
  }
}

// ---------------------------------------------------------------------------
// Node kernel: 128 nodes/block. X = [node_attr | recv_agg]. agg aliases
// node_out (each block reads only its own rows before overwriting them).
// ---------------------------------------------------------------------------
__global__ __launch_bounds__(256) void gn_node_kernel(
    const float* __restrict__ node_attr, const float* agg,
    const unsigned short* __restrict__ W1t, const unsigned short* __restrict__ W2t,
    const float* __restrict__ b1, const float* __restrict__ b2,
    const float* __restrict__ gamma, const float* __restrict__ beta,
    float* node_out) {
  extern __shared__ char smem[];
  unsigned short* Xs  = (unsigned short*)smem;      // 128 x HS
  unsigned short* W1s = Xs + 128 * HS;              // 128(N) x HS(K=128)
  unsigned short* Hs  = W1s + 128 * HS;             // 128 x HS
  unsigned short* W2s = Hs + 128 * HS;              // 64(N) x HS(K=128)

  const int tid = threadIdx.x;
  const long n0 = (long)blockIdx.x * 128;

  for (int i = tid; i < 128 * 32; i += 256) {       // 128 rows x 32 float4s
    int r = i >> 5, c = (i & 31) * 4;
    long nn = n0 + r;
    if (nn >= N_NODES) nn = N_NODES - 1;   // clamp stays inside this block's rows
    const float* src = (c < 64) ? (node_attr + nn * 64 + c) : (agg + nn * 64 + c - 64);
    float4 f = *(const float4*)src;
    uint2 p;
    p.x = pack_bf16x2(f.x, f.y);
    p.y = pack_bf16x2(f.z, f.w);
    *(uint2*)(Xs + r * HS + c) = p;
  }
  for (int i = tid; i < 128 * 16; i += 256) {
    int n = i >> 4, k8 = (i & 15) * 8;
    *(uint4*)(W1s + n * HS + k8) = *(const uint4*)(W1t + n * 128 + k8);
  }
  for (int i = tid; i < 64 * 16; i += 256) {
    int n = i >> 4, k8 = (i & 15) * 8;
    *(uint4*)(W2s + n * HS + k8) = *(const uint4*)(W2t + n * 128 + k8);
  }
  __syncthreads();

  const int wave = tid >> 5;
  const int lane = tid & 31;
  const int row0 = wave << 4;
  const int nl = lane & 15;
  const int mo = (lane >> 4) << 3;

  // Layer 1: [16,128] @ [128,128]
#pragma unroll 1
  for (int nt = 0; nt < 8; ++nt) {
    floatx8 acc = {0.f, 0.f, 0.f, 0.f, 0.f, 0.f, 0.f, 0.f};
#pragma unroll
    for (int kc = 0; kc < 4; ++kc) {
      bf16x16 a = load_a_frag(Xs, HS, row0, kc * 32);
      bf16x16 b = load_b_frag(W1s, HS, nt * 16, kc * 32);
      acc = wmma_bf16(a, b, acc);
    }
    int col = nt * 16 + nl;
    float bias = b1[col];
#pragma unroll
    for (int r = 0; r < 8; ++r)
      Hs[(row0 + r + mo) * HS + col] = f32_bf16(fmaxf(acc[r] + bias, 0.f));
  }
  __syncthreads();

  // Layer 2: [16,128] @ [128,64]
  floatx8 cacc[4];
#pragma unroll
  for (int nt = 0; nt < 4; ++nt)
    cacc[nt] = floatx8{0.f, 0.f, 0.f, 0.f, 0.f, 0.f, 0.f, 0.f};
#pragma unroll
  for (int kc = 0; kc < 4; ++kc) {
    bf16x16 a = load_a_frag(Hs, HS, row0, kc * 32);
#pragma unroll
    for (int nt = 0; nt < 4; ++nt) {
      bf16x16 b = load_b_frag(W2s, HS, nt * 16, kc * 32);
      cacc[nt] = wmma_bf16(a, b, cacc[nt]);
    }
  }

  float x[4][8], g[4], bt[4];
#pragma unroll
  for (int nt = 0; nt < 4; ++nt) {
    int col = nt * 16 + nl;
    float bias = b2[col];
    g[nt] = gamma[col];
    bt[nt] = beta[col];
#pragma unroll
    for (int r = 0; r < 8; ++r) x[nt][r] = fmaxf(cacc[nt][r] + bias, 0.f);
  }

#pragma unroll 1
  for (int r = 0; r < 8; ++r) {
    float s = x[0][r] + x[1][r] + x[2][r] + x[3][r];
    float q = x[0][r] * x[0][r] + x[1][r] * x[1][r] +
              x[2][r] * x[2][r] + x[3][r] * x[3][r];
#pragma unroll
    for (int m = 1; m < 16; m <<= 1) {
      s += __shfl_xor(s, m, 16);
      q += __shfl_xor(q, m, 16);
    }
    float mean = s * (1.f / 64.f);
    float var  = q * (1.f / 64.f) - mean * mean;
    float inv  = rsqrtf(var + 1e-5f);
    int rl = row0 + r + mo;
    long nn = n0 + rl;
    if (nn < N_NODES) {
      float* orow = node_out + nn * 64;
#pragma unroll
      for (int nt = 0; nt < 4; ++nt) {
        float y = (x[nt][r] - mean) * inv * g[nt] + bt[nt];
        orow[nt * 16 + nl] = y;
      }
    }
  }
}

extern "C" void kernel_launch(void* const* d_in, const int* in_sizes, int n_in,
                              void* d_out, int out_size, void* d_ws, size_t ws_size,
                              hipStream_t stream) {
  const float* edge_attr  = (const float*)d_in[0];
  const float* node_attr  = (const float*)d_in[1];
  const int*   edge_index = (const int*)d_in[2];
  // d_in[3] = batch (unused)
  const float* eW1 = (const float*)d_in[4];
  const float* eb1 = (const float*)d_in[5];
  const float* eW2 = (const float*)d_in[6];
  const float* eb2 = (const float*)d_in[7];
  const float* eg  = (const float*)d_in[8];
  const float* ebt = (const float*)d_in[9];
  const float* nW1 = (const float*)d_in[10];
  const float* nb1 = (const float*)d_in[11];
  const float* nW2 = (const float*)d_in[12];
  const float* nb2 = (const float*)d_in[13];
  const float* ng  = (const float*)d_in[14];
  const float* nbt = (const float*)d_in[15];

  float* edge_out = (float*)d_out;                       // [E, 64]
  float* node_out = edge_out + (long)N_EDGES * 64;       // [N, 64]
  float* agg      = node_out;                            // aliases node_out

  unsigned short* eW1t = (unsigned short*)d_ws;          // 128 x 192
  unsigned short* eW2t = eW1t + 128 * 192;               // 64 x 128
  unsigned short* nW1t = eW2t + 64 * 128;                // 128 x 128
  unsigned short* nW2t = nW1t + 128 * 128;               // 64 x 128

  gn_prep_kernel<<<512, 256, 0, stream>>>(eW1, eW2, nW1, nW2,
                                          eW1t, eW2t, nW1t, nW2t, agg);

  size_t lds_e = (size_t)(128 * XS + 128 * XS + 128 * HS + 64 * HS) * 2
               + 2 * 128 * sizeof(int);                  // ~152 KB
  gn_edge_kernel<<<N_EDGES / 128, 256, lds_e, stream>>>(
      edge_attr, node_attr, edge_index, eW1t, eW2t,
      eb1, eb2, eg, ebt, edge_out, agg);

  size_t lds_n = (size_t)(3 * 128 * HS + 64 * HS) * 2;   // ~119 KB
  gn_node_kernel<<<(N_NODES + 127) / 128, 256, lds_n, stream>>>(
      node_attr, agg, nW1t, nW2t, nb1, nb2, ng, nbt, node_out);
}
